// MotionEnhanceEstimator_25451976196130
// MI455X (gfx1250) — matrix operand order repaired and checked by reading, not compiled, not run on hardware
//
#include <hip/hip_runtime.h>
#include <hip/hip_bf16.h>
#include <math.h>

// MotionEnhanceEstimator for MI455X (gfx1250, wave32).
// Channel contractions use v_wmma_f32_16x16x32_f16 with pre-packed f16
// A-fragments (weights) and branchless B-fragment builds. Wave indices go
// through readfirstlane so all tensor bases live in SGPRs (SADDR-form VMEM,
// 32-bit per-lane offsets). FFTs are radix-2 256-pt in LDS.

typedef __attribute__((ext_vector_type(16))) _Float16 v16h;
typedef __attribute__((ext_vector_type(8)))  float    v8f;

// ---------------------------------------------------------------- helpers ---

// K index held by A-matrix element e for a lane in half `hi` (16-bit A 16x32).
__device__ __forceinline__ int a_kidx(int hi, int e) {
  return (hi << 4) + e;
}

__device__ __forceinline__ float act_apply(float v, int act) {
  if (act == 1) return v > 0.f ? v : 0.f;
  if (act == 2) return 1.f / (1.f + __expf(-v));
  return v;
}

// Wave-uniform flat wave index, forced into an SGPR so that all derived
// tensor base pointers are scalar (enables SADDR-form global loads).
__device__ __forceinline__ int wave_index_sgpr() {
  int wid = threadIdx.x >> 5;
  int gw = (int)(blockIdx.x * (blockDim.x >> 5) + wid);
  return __builtin_amdgcn_readfirstlane(gw);
}

// ------------------------------------------------------ weight pack kernels --
// Pack 1x1 weights (Cout x Cin, f32) into A-fragment order:
// wp[((m*KB + kb)*32 + lane)*16 + e], zero-padded to Mpad=16*MT, Kpad=32*KB.
__global__ void pack_w1x1_kernel(const float* __restrict__ w,
                                 _Float16* __restrict__ wp,
                                 int Cin, int Cout, int MT, int KB) {
  int tid = blockIdx.x * blockDim.x + threadIdx.x;
  int total = MT * KB * 512;
  if (tid >= total) return;
  int e = tid & 15;
  int lane = (tid >> 4) & 31;
  int rest = tid >> 9;
  int kb = rest % KB;
  int m  = rest / KB;
  int lh = lane & 15, hi = lane >> 4;
  int co = m * 16 + lh;
  int ci = kb * 32 + a_kidx(hi, e);
  float v = (co < Cout && ci < Cin) ? w[co * Cin + ci] : 0.f;
  wp[tid] = (_Float16)v;
}

// Pack 3x3 weights (Cout x 32 x 3 x 3, f32) into per-tap A-fragment order:
// wp[((k*MT + m)*32 + lane)*16 + e], tap k = i*3+j.
__global__ void pack_w3x3_kernel(const float* __restrict__ w,
                                 _Float16* __restrict__ wp,
                                 int Cout, int MT) {
  int tid = blockIdx.x * blockDim.x + threadIdx.x;
  int total = 9 * MT * 512;
  if (tid >= total) return;
  int e = tid & 15;
  int lane = (tid >> 4) & 31;
  int rest = tid >> 9;
  int m = rest % MT;
  int k = rest / MT;
  int i = k / 3, j = k % 3;
  int lh = lane & 15, hi = lane >> 4;
  int co = m * 16 + lh;
  int ci = a_kidx(hi, e);
  float v = (co < Cout) ? w[((co * 32 + ci) * 3 + i) * 3 + j] : 0.f;
  wp[tid] = (_Float16)v;
}

// ------------------------------------------------------------- layernorm ---

__global__ void ln_c_kernel(const float* __restrict__ x,
                            const float* __restrict__ g,
                            const float* __restrict__ b,
                            float* __restrict__ y,
                            int C, int HW, long total) {
  long idx = (long)blockIdx.x * blockDim.x + threadIdx.x;
  if (idx >= total) return;
  unsigned hw = (unsigned)(idx % HW);
  int n = (int)(idx / HW);
  const float* xp = x + (long)n * C * HW + hw;
  float s = 0.f, ss = 0.f;
  for (unsigned c = 0; c < (unsigned)C; ++c) {
    float v = xp[c * HW];
    s += v; ss += v * v;
  }
  float mu = s / (float)C;
  float var = ss / (float)C - mu * mu;
  float inv = rsqrtf(var + 1e-5f);
  float* yp = y + (long)n * C * HW + hw;
  for (unsigned c = 0; c < (unsigned)C; ++c)
    yp[c * HW] = (xp[c * HW] - mu) * inv * g[c] + b[c];
}

// ------------------------------------------------------- 1x1 conv (WMMA) ---
// One wave: 16 positions x all Cout. All KB B-fragments loaded up front; each
// M-tile's accumulator lives only for its own K-loop + store (no spills).
template <int Cin, int Cout>
__global__ void conv1x1_wmma_kernel(const float* __restrict__ x,
                                    const _Float16* __restrict__ wp,
                                    const float* __restrict__ bias,
                                    const float* __restrict__ residual,
                                    float* __restrict__ out,
                                    int HW, int P, int act) {
  constexpr int MT = Cout / 16;
  constexpr int KB = Cin / 32;
  const int lane = threadIdx.x & 31;
  const int gw = wave_index_sgpr();           // SGPR wave index
  if (gw >= (P >> 4)) return;                 // SALU guard (EXEC all-1)
  const int p0  = gw << 4;
  const int n   = p0 / HW;                    // scalar
  const int hw0 = p0 % HW;                    // scalar
  const int lh = lane & 15;
  const int hi = lane >> 4;
  const unsigned uHW = (unsigned)HW;
  // scalar bases + unsigned 32-bit per-lane offsets (SADDR-form loads)
  const float* xcol = x + (long)n * Cin * HW + hw0;
  float* ocol = out + (long)n * Cout * HW + hw0;
  const float* rcol = residual ? residual + (long)n * Cout * HW + hw0
                               : nullptr;

  v16h b[KB];
#pragma unroll
  for (int kb = 0; kb < KB; ++kb) {
    float bv[16];
#pragma unroll
    for (int e = 0; e < 16; ++e)
      bv[e] = xcol[(unsigned)(kb * 32 + (hi << 4) + e) * uHW + (unsigned)lh];
#pragma unroll
    for (int e = 0; e < 16; ++e) b[kb][e] = (_Float16)bv[e];
  }
#pragma unroll
  for (int m = 0; m < MT; ++m) {
    v8f acc = {};
#pragma unroll
    for (int kb = 0; kb < KB; ++kb) {
      const v16h a = *(const v16h*)(wp + (((m * KB + kb) * 32) + lane) * 16);
      acc = __builtin_amdgcn_wmma_f32_16x16x32_f16(false, a, false, b[kb],
                                                   (short)0, acc, false, false);
    }
#pragma unroll
    for (int r = 0; r < 8; ++r) {
      unsigned co = (unsigned)(m * 16 + r + (hi << 3));
      float v = acc[r] + bias[co];
      v = act_apply(v, act);
      if (rcol) v += rcol[co * uHW + (unsigned)lh];
      ocol[co * uHW + (unsigned)lh] = v;
    }
  }
}

// ----------------------------------------------- full 3x3 conv (WMMA, Cin=32)
template <int Cout>
__global__ void conv3x3_wmma_kernel(const float* __restrict__ x,
                                    const _Float16* __restrict__ wp,
                                    const float* __restrict__ bias,
                                    float* __restrict__ out,
                                    int H, int W, int P, int act) {
  constexpr int MT = (Cout + 15) / 16;
  const int lane = threadIdx.x & 31;
  const int HW = H * W;
  const unsigned uHW = (unsigned)HW;
  const int gw = wave_index_sgpr();
  if (gw >= (P >> 4)) return;
  const int p0  = gw << 4;
  const int n   = p0 / HW;
  const int hw0 = p0 % HW;
  const int lh = lane & 15;
  const int hi = lane >> 4;
  const int hw = hw0 + lh;
  const int h = hw / W;
  const int wc = hw % W;
  const float* xb = x + (long)n * 32 * HW;     // scalar base
  float* ob = out + (long)n * Cout * HW + hw0;

  v8f acc[MT] = {};
#pragma unroll
  for (int k = 0; k < 9; ++k) {
    const int i = k / 3, j = k % 3;
    int hh = h + i - 1, ws = wc + j - 1;
    float inb = (hh >= 0 && hh < H && ws >= 0 && ws < W) ? 1.f : 0.f;
    int hc  = min(max(hh, 0), H - 1);
    int wcl = min(max(ws, 0), W - 1);
    unsigned sb = (unsigned)(hc * W + wcl);     // unsigned per-lane offset
    float bv[16];
#pragma unroll
    for (int e = 0; e < 16; ++e)
      bv[e] = xb[(unsigned)((hi << 4) + e) * uHW + sb];
    v16h b;
#pragma unroll
    for (int e = 0; e < 16; ++e) b[e] = (_Float16)(bv[e] * inb);
#pragma unroll
    for (int m = 0; m < MT; ++m) {
      const v16h a = *(const v16h*)(wp + ((k * MT + m) * 32 + lane) * 16);
      acc[m] = __builtin_amdgcn_wmma_f32_16x16x32_f16(false, a, false, b,
                                                      (short)0, acc[m],
                                                      false, false);
    }
  }
#pragma unroll
  for (int m = 0; m < MT; ++m) {
#pragma unroll
    for (int r = 0; r < 8; ++r) {
      int coo = m * 16 + r + (hi << 3);
      if (coo < Cout) {
        float v = acc[m][r] + bias[coo];
        ob[(unsigned)coo * uHW + (unsigned)lh] = act_apply(v, act);
      }
    }
  }
}

// ------------------------------------------------------ depthwise 3x3 conv --
__global__ void dw3x3_kernel(const float* __restrict__ x,
                             const float* __restrict__ w,
                             const float* __restrict__ b,
                             float* __restrict__ y,
                             int C, int H, int W, long total) {
  long idx = (long)blockIdx.x * blockDim.x + threadIdx.x;
  if (idx >= total) return;
  int wc = (int)(idx % W); long r = idx / W;
  int h  = (int)(r % H);   r /= H;
  int c  = (int)(r % C);
  int n  = (int)(r / C);
  const int HW = H * W;
  const float* xp = x + ((long)n * C + c) * HW;
  float acc = b[c];
#pragma unroll
  for (int i = 0; i < 3; ++i)
#pragma unroll
    for (int j = 0; j < 3; ++j) {
      int hh = h + i - 1, ws = wc + j - 1;
      float inb = (hh >= 0 && hh < H && ws >= 0 && ws < W) ? 1.f : 0.f;
      int hc  = min(max(hh, 0), H - 1);
      int wcl = min(max(ws, 0), W - 1);
      acc += w[(c * 3 + i) * 3 + j] * xp[(unsigned)(hc * W + wcl)] * inb;
    }
  y[idx] = acc;
}

// ------------------------------------------- modulated deform conv (WMMA) ---
__global__ void deform_wmma_kernel(const float* __restrict__ x,
                                   const float* __restrict__ offset,
                                   const float* __restrict__ mask,
                                   const _Float16* __restrict__ wp, // packed
                                   float* __restrict__ out,
                                   int H, int W, int P) {
  const int lane = threadIdx.x & 31;
  const int HW = H * W;
  const unsigned uHW = (unsigned)HW;
  const int gw = wave_index_sgpr();
  if (gw >= (P >> 4)) return;
  const int p0  = gw << 4;
  const int n   = p0 / HW;
  const int hw0 = p0 % HW;
  const int lh = lane & 15;
  const int hi = lane >> 4;
  const unsigned hw = (unsigned)(hw0 + lh);
  const int h = (int)(hw / (unsigned)W), wc = (int)(hw % (unsigned)W);
  // scalar bases, unsigned 32-bit per-lane offsets (SADDR-form loads)
  const float* xb  = x + (long)n * 32 * HW;
  const float* ofb = offset + (long)n * 18 * HW;
  const float* mkb = mask + (long)n * 9 * HW;
  float* ob = out + (long)n * 32 * HW + hw0;

  v8f acc0 = {}, acc1 = {};
#pragma unroll
  for (int k = 0; k < 9; ++k) {
    const int i = k / 3, j = k % 3;
    float oy = ofb[(unsigned)(2 * k) * uHW + hw];
    float ox = ofb[(unsigned)(2 * k + 1) * uHW + hw];
    float m  = mkb[(unsigned)k * uHW + hw];
    float py = (float)h + (float)(i - 1) + oy;
    float px = (float)wc + (float)(j - 1) + ox;
    float fy = floorf(py), fx = floorf(px);
    float wy1 = py - fy, wx1 = px - fx;
    float wy0 = 1.f - wy1, wx0 = 1.f - wx1;
    int y0 = (int)fy, x0 = (int)fx;
    float cw[4]; unsigned caddr[4];
    int   ys[2] = {y0, y0 + 1}, xs[2] = {x0, x0 + 1};
    float wy[2] = {wy0, wy1},   wx[2] = {wx0, wx1};
#pragma unroll
    for (int a2 = 0; a2 < 2; ++a2)
#pragma unroll
      for (int b2 = 0; b2 < 2; ++b2) {
        int t = a2 * 2 + b2;
        int yy = ys[a2], xx = xs[b2];
        bool valid = (yy >= 0) && (yy <= H - 1) && (xx >= 0) && (xx <= W - 1);
        int yc = min(max(yy, 0), H - 1), xc = min(max(xx, 0), W - 1);
        cw[t] = valid ? (wy[a2] * wx[b2] * m) : 0.f;
        caddr[t] = (unsigned)(yc * W + xc);
      }
    v16h b;
#pragma unroll
    for (int e = 0; e < 16; ++e) {
      unsigned cb = (unsigned)((hi << 4) + e) * uHW;
      float v = cw[0] * xb[cb + caddr[0]] + cw[1] * xb[cb + caddr[1]] +
                cw[2] * xb[cb + caddr[2]] + cw[3] * xb[cb + caddr[3]];
      b[e] = (_Float16)v;
    }
    const v16h a0 = *(const v16h*)(wp + ((k * 2 + 0) * 32 + lane) * 16);
    const v16h a1 = *(const v16h*)(wp + ((k * 2 + 1) * 32 + lane) * 16);
    acc0 = __builtin_amdgcn_wmma_f32_16x16x32_f16(false, a0, false, b,
                                                  (short)0, acc0, false, false);
    acc1 = __builtin_amdgcn_wmma_f32_16x16x32_f16(false, a1, false, b,
                                                  (short)0, acc1, false, false);
  }
#pragma unroll
  for (int r = 0; r < 8; ++r) {
    unsigned c0 = (unsigned)(r + (hi << 3));
    ob[c0 * uHW + (unsigned)lh]        = acc0[r];
    ob[(16 + c0) * uHW + (unsigned)lh] = acc1[r];
  }
}

// ------------------------------------------------------- 256-pt FFT in LDS --
__device__ __forceinline__ void fft256(float* sRe, float* sIm, int t, float sign) {
  for (int s = 1; s <= 8; ++s) {
    int m = 1 << s, half = m >> 1;
    int j = t & (half - 1);
    int i0 = ((t >> (s - 1)) << s) + j;
    int i1 = i0 + half;
    float ang = sign * 6.28318530717958647692f * (float)j / (float)m;
    float cwv = __cosf(ang), swv = __sinf(ang);
    __syncthreads();
    float ur = sRe[i0], ui = sIm[i0];
    float vr = sRe[i1], vi = sIm[i1];
    float tr = vr * cwv - vi * swv;
    float ti = vr * swv + vi * cwv;
    sRe[i0] = ur + tr; sIm[i0] = ui + ti;
    sRe[i1] = ur - tr; sIm[i1] = ui - ti;
  }
  __syncthreads();
}

// rfft along W; writes into split-plane Fbuf (N, 2C, H, Wf): re@c, im@C+c.
__global__ void rfft_rows_kernel(const float* __restrict__ x,
                                 float* __restrict__ F,
                                 int C, int H, int W, int Wf) {
  __shared__ float sRe[256], sIm[256];
  int rid = blockIdx.x;            // over N*C*H
  int t = threadIdx.x;             // 128 threads
  const float* row = x + (long)rid * W;
  for (int p = t; p < 256; p += 128) {
    int src = __brev(p) >> 24;
    sRe[p] = row[src];
    sIm[p] = 0.f;
  }
  fft256(sRe, sIm, t, -1.f);
  int h = rid % H; int c = (rid / H) % C; int n = rid / (H * C);
  long reB = (((long)n * (2 * C) + c) * H + h) * Wf;
  long imB = (((long)n * (2 * C) + C + c) * H + h) * Wf;
  for (int k = t; k < Wf; k += 128) {
    F[reB + k] = sRe[k];
    F[imB + k] = sIm[k];
  }
}

// complex FFT along H, in place on split-plane Fbuf.
__global__ void fft_cols_kernel(float* __restrict__ F, int C, int H, int Wf) {
  __shared__ float sRe[256], sIm[256];
  int cid = blockIdx.x;            // over N*C*Wf
  int t = threadIdx.x;
  int wf = cid % Wf; int c = (cid / Wf) % C; int n = cid / (Wf * C);
  long reB = (((long)n * (2 * C) + c) * H) * Wf + wf;
  long imB = (((long)n * (2 * C) + C + c) * H) * Wf + wf;
  for (int p = t; p < 256; p += 128) {
    unsigned src = __brev(p) >> 24;
    sRe[p] = F[reB + src * (unsigned)Wf];
    sIm[p] = F[imB + src * (unsigned)Wf];
  }
  fft256(sRe, sIm, t, -1.f);
  for (unsigned hh = t; hh < 256; hh += 128) {
    F[reB + hh * (unsigned)Wf] = sRe[hh];
    F[imB + hh * (unsigned)Wf] = sIm[hh];
  }
}

// inverse FFT along H with fused real spectral-mask multiply; scale 1/H.
__global__ void ifft_cols_mask_kernel(float* __restrict__ F,
                                      const float* __restrict__ mask,
                                      int C, int H, int Wf) {
  __shared__ float sRe[256], sIm[256];
  int cid = blockIdx.x;
  int t = threadIdx.x;
  int wf = cid % Wf; int c = (cid / Wf) % C; int n = cid / (Wf * C);
  long reB = (((long)n * (2 * C) + c) * H) * Wf + wf;
  long imB = (((long)n * (2 * C) + C + c) * H) * Wf + wf;
  long mB  = (((long)n * C + c) * H) * Wf + wf;
  for (int p = t; p < 256; p += 128) {
    unsigned src = __brev(p) >> 24;
    float mm = mask[mB + src * (unsigned)Wf];
    sRe[p] = F[reB + src * (unsigned)Wf] * mm;
    sIm[p] = F[imB + src * (unsigned)Wf] * mm;
  }
  fft256(sRe, sIm, t, 1.f);
  const float sc = 1.f / 256.f;
  for (unsigned hh = t; hh < 256; hh += 128) {
    F[reB + hh * (unsigned)Wf] = sRe[hh] * sc;
    F[imB + hh * (unsigned)Wf] = sIm[hh] * sc;
  }
}

// inverse rfft along W (Hermitian reconstruction); writes real, scale 1/W.
__global__ void irfft_rows_kernel(const float* __restrict__ F,
                                  float* __restrict__ out,
                                  int C, int H, int W, int Wf) {
  __shared__ float sRe[256], sIm[256];
  int rid = blockIdx.x;            // over N*C*H
  int t = threadIdx.x;
  int h = rid % H; int c = (rid / H) % C; int n = rid / (H * C);
  long reB = (((long)n * (2 * C) + c) * H + h) * Wf;
  long imB = (((long)n * (2 * C) + C + c) * H + h) * Wf;
  for (int p = t; p < 256; p += 128) {
    int k = __brev(p) >> 24;
    int neg = (k >= Wf);
    unsigned kk = (unsigned)(neg ? (256 - k) : k);   // branchless index select
    float sgn = neg ? -1.f : 1.f;
    sRe[p] = F[reB + kk];
    sIm[p] = F[imB + kk] * sgn;
  }
  fft256(sRe, sIm, t, 1.f);
  const float sc = 1.f / 256.f;
  long ob = (long)rid * W;
  for (int p = t; p < 256; p += 128) out[ob + p] = sRe[p] * sc;
}

// ------------------------------- fused depthwise-3x3 + gated GELU (fl_w1) ---
// Writes a 32-channel buffer: channel c = a*gelu(b), channel c+16 = 0
// (zero pad so the final 16->32 conv can run with Cin=32, K zero-padded).
__global__ void dwgate_kernel(const float* __restrict__ x,
                              const float* __restrict__ w,
                              const float* __restrict__ bias,
                              float* __restrict__ y,
                              int H, int W, long total) {   // total = N*16*H*W
  long idx = (long)blockIdx.x * blockDim.x + threadIdx.x;
  if (idx >= total) return;
  int wc = (int)(idx % W); long r = idx / W;
  int h  = (int)(r % H);   r /= H;
  int c  = (int)(r % 16);
  int n  = (int)(r / 16);
  const int HW = H * W;
  const float* xa = x + ((long)n * 32 + c) * HW;
  const float* xb = x + ((long)n * 32 + c + 16) * HW;
  float a = bias[c], bb = bias[c + 16];
#pragma unroll
  for (int i = 0; i < 3; ++i)
#pragma unroll
    for (int j = 0; j < 3; ++j) {
      int hh = h + i - 1, ws = wc + j - 1;
      float inb = (hh >= 0 && hh < H && ws >= 0 && ws < W) ? 1.f : 0.f;
      int hc  = min(max(hh, 0), H - 1);
      int wcl = min(max(ws, 0), W - 1);
      unsigned s = (unsigned)(hc * W + wcl);
      a  += w[(c * 3 + i) * 3 + j]        * xa[s] * inb;
      bb += w[((c + 16) * 3 + i) * 3 + j] * xb[s] * inb;
    }
  float g = 0.5f * bb * (1.f + erff(bb * 0.70710678118654752440f));
  long ob = ((long)n * 32 + c) * HW + (long)h * W + wc;
  y[ob] = a * g;
  y[ob + (long)16 * HW] = 0.f;   // zero pad channels 16..31
}

// ------------------------------------------------------------------ launch --

extern "C" void kernel_launch(void* const* d_in, const int* in_sizes, int n_in,
                              void* d_out, int out_size, void* d_ws, size_t ws_size,
                              hipStream_t stream) {
  (void)in_sizes; (void)n_in; (void)out_size; (void)ws_size;
  const float* input_flow  = (const float*)d_in[0];
  const float* input_event = (const float*)d_in[1];
  const float* ln_flow_g   = (const float*)d_in[2];
  const float* ln_flow_b   = (const float*)d_in[3];
  const float* ln_event_g  = (const float*)d_in[4];
  const float* ln_event_b  = (const float*)d_in[5];
  const float* ff_w1 = (const float*)d_in[6];  const float* ff_b1 = (const float*)d_in[7];
  const float* ff_w2 = (const float*)d_in[8];  const float* ff_b2 = (const float*)d_in[9];
  const float* fo_w1 = (const float*)d_in[10]; const float* fo_b1 = (const float*)d_in[11];
  const float* fo_w2 = (const float*)d_in[12]; const float* fo_b2 = (const float*)d_in[13];
  const float* fm_w1 = (const float*)d_in[14]; const float* fm_b1 = (const float*)d_in[15];
  const float* fm_w2 = (const float*)d_in[16]; const float* fm_b2 = (const float*)d_in[17];
  const float* dc_w  = (const float*)d_in[18];
  const float* sf_w1 = (const float*)d_in[19]; const float* sf_b1 = (const float*)d_in[20];
  const float* sf_w2 = (const float*)d_in[21]; const float* sf_b2 = (const float*)d_in[22];
  const float* sf_w3 = (const float*)d_in[23]; const float* sf_b3 = (const float*)d_in[24];
  const float* sf_w4 = (const float*)d_in[25]; const float* sf_b4 = (const float*)d_in[26];
  const float* fl_w1 = (const float*)d_in[27]; const float* fl_b1 = (const float*)d_in[28];
  const float* fl_w2 = (const float*)d_in[29]; const float* fl_b2 = (const float*)d_in[30];
  float* outp = (float*)d_out;

  const int  N = 8, C = 32, H = 256, W = 256, Wf = 129;
  const int  HW  = H * W;                      // 65536
  const int  HWf = H * Wf;                     // 33024
  const long S    = (long)N * C * HW;          // 16,777,216
  const long S18  = (long)N * 18 * HW;
  const long S9   = (long)N * 9 * HW;
  const long SF64 = (long)N * 64 * HWf;        // split re/im planes
  const long SPM  = (long)N * 32 * HWf;
  const int  P1   = N * HW;                    // spatial positions
  const int  P2   = N * HWf;                   // spectral positions

  float* ws  = (float*)d_ws;
  float* fn  = ws;              // flow_norm   [S]
  float* en  = fn  + S;         // event_norm  [S]
  float* tmp = en  + S;         // 1x1 scratch [S]
  float* xb  = tmp + S;         // x           [S]
  float* off = xb  + S;         // offset      [S18]
  float* msk = off + S18;       // mask        [S9]
  float* dc  = msk + S9;        // dc          [S]
  float* xf  = dc  + S;         // x_fft       [SF64]
  float* df  = xf  + SF64;      // dc_fft      [SF64]
  float* h1  = df  + SF64;      // spectral h  [SF64]
  float* h2  = h1  + SF64;      // spectral h  [SF64]
  float* spm = h2  + SF64;      // sp_mask     [SPM]
  float* xi  = fn;              // reuse: x_irfft         [S]
  float* gg  = en;              // reuse: gated (N,32,HW) [S]

  // packed f16 weight arena (after spm; 32B-aligned offsets)
  _Float16* wp0 = (_Float16*)(spm + SPM);
  _Float16* pw_ff1 = wp0;                 // MT2 KB1: 1024
  _Float16* pw_fo1 = pw_ff1 + 1024;
  _Float16* pw_fm1 = pw_fo1 + 1024;
  _Float16* pw_sf1 = pw_fm1 + 1024;       // MT4 KB2: 4096
  _Float16* pw_sf3 = pw_sf1 + 4096;
  _Float16* pw_sf4 = pw_sf3 + 4096;       // MT2 KB2: 2048
  _Float16* pw_fl2 = pw_sf4 + 2048;       // MT2 KB1: 1024 (Cin=16 zero-padded)
  _Float16* pw_fo2 = pw_fl2 + 1024;       // 9*MT2*512: 9216
  _Float16* pw_fm2 = pw_fo2 + 9216;       // 9*MT1*512: 4608
  _Float16* pw_dc  = pw_fm2 + 4608;       // 9*MT2*512: 9216

  const int BLK = 256;
  auto blocks  = [](long total, int blk) { return (unsigned)((total + blk - 1) / blk); };
  auto wblocks = [](long waves) { return (unsigned)((waves + 7) / 8); };  // 8 waves/block

  // 0) pack weights into A-fragment f16 layout (tiny, L2-resident)
  pack_w1x1_kernel<<<blocks(2 * 1 * 512, BLK), BLK, 0, stream>>>(ff_w1, pw_ff1, 32, 32, 2, 1);
  pack_w1x1_kernel<<<blocks(2 * 1 * 512, BLK), BLK, 0, stream>>>(fo_w1, pw_fo1, 32, 32, 2, 1);
  pack_w1x1_kernel<<<blocks(2 * 1 * 512, BLK), BLK, 0, stream>>>(fm_w1, pw_fm1, 32, 32, 2, 1);
  pack_w1x1_kernel<<<blocks(4 * 2 * 512, BLK), BLK, 0, stream>>>(sf_w1, pw_sf1, 64, 64, 4, 2);
  pack_w1x1_kernel<<<blocks(4 * 2 * 512, BLK), BLK, 0, stream>>>(sf_w3, pw_sf3, 64, 64, 4, 2);
  pack_w1x1_kernel<<<blocks(2 * 2 * 512, BLK), BLK, 0, stream>>>(sf_w4, pw_sf4, 64, 32, 2, 2);
  pack_w1x1_kernel<<<blocks(2 * 1 * 512, BLK), BLK, 0, stream>>>(fl_w2, pw_fl2, 16, 32, 2, 1);
  pack_w3x3_kernel<<<blocks(9 * 2 * 512, BLK), BLK, 0, stream>>>(fo_w2, pw_fo2, 18, 2);
  pack_w3x3_kernel<<<blocks(9 * 1 * 512, BLK), BLK, 0, stream>>>(fm_w2, pw_fm2, 9, 1);
  pack_w3x3_kernel<<<blocks(9 * 2 * 512, BLK), BLK, 0, stream>>>(dc_w,  pw_dc, 32, 2);

  // 1) LayerNorm over channels
  ln_c_kernel<<<blocks(P1, BLK), BLK, 0, stream>>>(input_flow,  ln_flow_g,  ln_flow_b,  fn, C, HW, (long)P1);
  ln_c_kernel<<<blocks(P1, BLK), BLK, 0, stream>>>(input_event, ln_event_g, ln_event_b, en, C, HW, (long)P1);

  // 2) x = dw3x3( relu(conv1x1(flow_norm)) )
  conv1x1_wmma_kernel<32, 32><<<wblocks(P1 >> 4), BLK, 0, stream>>>(fn, pw_ff1, ff_b1, nullptr, tmp, HW, P1, 1);
  dw3x3_kernel<<<blocks(S, BLK), BLK, 0, stream>>>(tmp, ff_w2, ff_b2, xb, C, H, W, S);

  // 3) offset = conv3x3( relu(conv1x1(event_norm)) ), C->18
  conv1x1_wmma_kernel<32, 32><<<wblocks(P1 >> 4), BLK, 0, stream>>>(en, pw_fo1, fo_b1, nullptr, tmp, HW, P1, 1);
  conv3x3_wmma_kernel<18><<<wblocks(P1 >> 4), BLK, 0, stream>>>(tmp, pw_fo2, fo_b2, off, H, W, P1, 0);

  // 4) mask = sigmoid(conv3x3( relu(conv1x1(event_norm)) )), C->9
  conv1x1_wmma_kernel<32, 32><<<wblocks(P1 >> 4), BLK, 0, stream>>>(en, pw_fm1, fm_b1, nullptr, tmp, HW, P1, 1);
  conv3x3_wmma_kernel<9><<<wblocks(P1 >> 4), BLK, 0, stream>>>(tmp, pw_fm2, fm_b2, msk, H, W, P1, 2);

  // 5) deformable conv
  deform_wmma_kernel<<<wblocks(P1 >> 4), BLK, 0, stream>>>(xb, off, msk, pw_dc, dc, H, W, P1);

  // 6) rfft2 of x and dc -> split-plane (N,64,H,Wf)
  rfft_rows_kernel<<<(unsigned)(N * C * H), 128, 0, stream>>>(xb, xf, C, H, W, Wf);
  fft_cols_kernel<<<(unsigned)(N * C * Wf), 128, 0, stream>>>(xf, C, H, Wf);
  rfft_rows_kernel<<<(unsigned)(N * C * H), 128, 0, stream>>>(dc, df, C, H, W, Wf);
  fft_cols_kernel<<<(unsigned)(N * C * Wf), 128, 0, stream>>>(df, C, H, Wf);

  // 7) spectral mask path on concat(real,imag) = df as (N,64,H,Wf)
  conv1x1_wmma_kernel<64, 64><<<wblocks(P2 >> 4), BLK, 0, stream>>>(df, pw_sf1, sf_b1, nullptr, h1, HWf, P2, 0);
  dw3x3_kernel<<<blocks(SF64, BLK), BLK, 0, stream>>>(h1, sf_w2, sf_b2, h2, 64, H, Wf, SF64);
  conv1x1_wmma_kernel<64, 64><<<wblocks(P2 >> 4), BLK, 0, stream>>>(h2, pw_sf3, sf_b3, nullptr, h1, HWf, P2, 1);
  conv1x1_wmma_kernel<64, 32><<<wblocks(P2 >> 4), BLK, 0, stream>>>(h1, pw_sf4, sf_b4, nullptr, spm, HWf, P2, 2);

  // 8) x_irfft = irfft2( x_fft * sp_mask )
  ifft_cols_mask_kernel<<<(unsigned)(N * C * Wf), 128, 0, stream>>>(xf, spm, C, H, Wf);
  irfft_rows_kernel<<<(unsigned)(N * C * H), 128, 0, stream>>>(xf, xi, C, H, W, Wf);

  // 9) gated depthwise: g = a * gelu(b), zero-padded to 32 channels
  dwgate_kernel<<<blocks((long)N * 16 * HW, BLK), BLK, 0, stream>>>(xi, fl_w1, fl_b1, gg, H, W, (long)N * 16 * HW);

  // 10) out = conv1x1(g, 16(pad 32)->32) + input_flow
  conv1x1_wmma_kernel<32, 32><<<wblocks(P1 >> 4), BLK, 0, stream>>>(gg, pw_fl2, fl_b2, input_flow, outp, HW, P1, 0);
}
